// MP_MobileNetV3Controller_84250078478432
// MI455X (gfx1250) — compile-verified, matrix operand order checked by reading, not run on hardware
//
#include <hip/hip_runtime.h>
#include <math.h>

// ---------------------------------------------------------------------------
// MP MobileNetV3 controller on gfx1250 (MI455X).
// B=4096, H=1024, 45 sequential LSTM steps.
// Core: 44 x (h[4096,1024] @ W_hh.T[1024,4096]) via v_wmma_f32_16x16x32_bf16,
// with the x@W_ih.T GEMM eliminated algebraically (embeds come from 3-row
// tables -> precomputed gate rows + per-row gather).
// ---------------------------------------------------------------------------

#define B_SZ   4096
#define H_SZ   1024
#define FH_SZ  4096   // 4*H

typedef __attribute__((ext_vector_type(16))) __bf16 v16bf;
typedef __attribute__((ext_vector_type(8)))  float  v8f;

union Frag { uint4 q[2]; v16bf v; };

__device__ __forceinline__ unsigned short f2bf(float x) {
  unsigned int u = __float_as_uint(x);
  u += 0x7FFFu + ((u >> 16) & 1u);          // round-to-nearest-even
  return (unsigned short)(u >> 16);
}

__device__ __forceinline__ float sigmoidf_(float x) {
  return 1.0f / (1.0f + __expf(-x));
}

// --------------------------- precompute kernels -----------------------------

// W_hh fp32 -> bf16 (row-major [4096][1024])
__global__ __launch_bounds__(256) void convert_bf16_kernel(
    const float* __restrict__ src, unsigned short* __restrict__ dst) {
  int i = blockIdx.x * 256 + threadIdx.x;
  dst[i] = f2bf(src[i]);
}

// Q[9][4096]: rows 0-2 depth, 3-5 ratio, 6-8 ks.
// Q[r][j] = emb_row . W_ih[j,:] + b_ih[j] + b_hh[j]
__global__ __launch_bounds__(256) void precompute_q_kernel(
    const float* __restrict__ depth_emb, const float* __restrict__ ratio_emb,
    const float* __restrict__ ks_emb, const float* __restrict__ W_ih,
    const float* __restrict__ b_ih, const float* __restrict__ b_hh,
    float* __restrict__ Q) {
  int idx = blockIdx.x * 256 + threadIdx.x;      // < 9*4096
  int r = idx >> 12, j = idx & 4095;
  const float* emb = (r < 3) ? (depth_emb + (size_t)r * H_SZ)
                   : (r < 6) ? (ratio_emb + (size_t)(r - 3) * H_SZ)
                             : (ks_emb    + (size_t)(r - 6) * H_SZ);
  const float* wr = W_ih + (size_t)j * H_SZ;
  float s = b_ih[j] + b_hh[j];
  for (int k = 0; k < H_SZ; ++k) s += emb[k] * wr[k];
  Q[idx] = s;
}

// C5[5][4096]: cond_emb row r (H/2) . W_ih[j, 0:512]
__global__ __launch_bounds__(256) void precompute_c5_kernel(
    const float* __restrict__ cond_emb, const float* __restrict__ W_ih,
    float* __restrict__ C5) {
  int idx = blockIdx.x * 256 + threadIdx.x;      // < 5*4096
  int r = idx >> 12, j = idx & 4095;
  const float* emb = cond_emb + (size_t)r * (H_SZ / 2);
  const float* wr  = W_ih + (size_t)j * H_SZ;
  float s = 0.f;
  for (int k = 0; k < H_SZ / 2; ++k) s += emb[k] * wr[k];
  C5[idx] = s;
}

// S[4096]: sc_emb[scid] . W_ih[j, 512:1024] + b_ih[j] + b_hh[j]
__global__ __launch_bounds__(256) void precompute_s_kernel(
    const float* __restrict__ sc_emb, const int* __restrict__ scid_p,
    const float* __restrict__ W_ih, const float* __restrict__ b_ih,
    const float* __restrict__ b_hh, float* __restrict__ S) {
  int j = blockIdx.x * 256 + threadIdx.x;        // < 4096
  int scid = *scid_p;
  const float* emb = sc_emb + (size_t)scid * (H_SZ / 2);
  const float* wr  = W_ih + (size_t)j * H_SZ + H_SZ / 2;
  float s = b_ih[j] + b_hh[j];
  for (int k = 0; k < H_SZ / 2; ++k) s += emb[k] * wr[k];
  S[j] = s;
}

// ------------------------------ step 0 --------------------------------------
// h0=c0=0, gates0 = w*C5[idx] + (1-w)*C5[idx+1] + S  (no GEMM at all)
__global__ __launch_bounds__(256) void lstm_step0_kernel(
    const float* __restrict__ constraints, const float* __restrict__ C5,
    const float* __restrict__ S, float* __restrict__ c_state,
    float* __restrict__ h_f32, unsigned short* __restrict__ h_bf) {
  int idx = blockIdx.x * 256 + threadIdx.x;      // < 4096*1024
  int b = idx >> 10, j = idx & 1023;
  float cst = constraints[b];
  float t = (cst - 10.0f) * 0.4f;                // / 2.5
  int ci = (int)floorf(t);
  ci = ci < 0 ? 0 : (ci > 3 ? 3 : ci);
  float left = 10.0f + 2.5f * (float)ci;
  float w = (left + 2.5f - cst) * 0.4f;
  float g4[4];
#pragma unroll
  for (int gi = 0; gi < 4; ++gi) {
    int col = gi * H_SZ + j;
    g4[gi] = w * C5[(size_t)ci * FH_SZ + col] +
             (1.0f - w) * C5[(size_t)(ci + 1) * FH_SZ + col] + S[col];
  }
  float c2 = sigmoidf_(g4[0]) * tanhf(g4[2]);    // f-term * c0 == 0
  float h2 = sigmoidf_(g4[3]) * tanhf(c2);
  size_t off = (size_t)b * H_SZ + j;
  c_state[off] = c2;
  h_f32[off] = h2;
  h_bf[off] = f2bf(h2);
}

// --------------------- fused WMMA GEMM + LSTM cell --------------------------
// gates = h @ W_hh.T + Q[s[b]]  ;  cell math in registers.
// Wave tile: 32(M) x 16(N) x 4 gates. Block = 8 waves. Grid = 1024 blocks.
__global__ __launch_bounds__(256) void lstm_step_kernel(
    const unsigned short* __restrict__ h_in,   // [4096][1024] bf16
    const unsigned short* __restrict__ whh,    // [4096][1024] bf16
    const float* __restrict__ qtab,            // [3][4096]
    const int* __restrict__ sidx,              // [4096]
    float* __restrict__ c_state,               // [4096][1024]
    float* __restrict__ h_out_f32,             // [4096][1024]
    unsigned short* __restrict__ h_out_bf) {   // [4096][1024]
  const int lane = threadIdx.x & 31;
  const int wid  = threadIdx.x >> 5;
  const int g    = blockIdx.x * 8 + wid;       // 8192 wave tiles
  const int m0   = (g >> 6) * 32;              // 128 m-tiles of 32 rows
  const int n0   = (g & 63) * 16;              // 64 n-tiles of 16 cols
  const int r    = lane & 15;
  const int hiK  = (lane >> 4) * 8;            // K sub-offset per half-wave

  v8f acc[2][4];
#pragma unroll
  for (int ms = 0; ms < 2; ++ms)
#pragma unroll
    for (int gi = 0; gi < 4; ++gi)
#pragma unroll
      for (int e = 0; e < 8; ++e) acc[ms][gi][e] = 0.0f;

  for (int k0 = 0; k0 < H_SZ; k0 += 32) {
    Frag A[2], Bf[4];
#pragma unroll
    for (int ms = 0; ms < 2; ++ms) {
      const unsigned short* p =
          h_in + (size_t)(m0 + ms * 16 + r) * H_SZ + k0 + hiK;
      A[ms].q[0] = *reinterpret_cast<const uint4*>(p);
      A[ms].q[1] = *reinterpret_cast<const uint4*>(p + 16);
    }
#pragma unroll
    for (int gi = 0; gi < 4; ++gi) {
      const unsigned short* p =
          whh + (size_t)(gi * H_SZ + n0 + r) * H_SZ + k0 + hiK;
      Bf[gi].q[0] = *reinterpret_cast<const uint4*>(p);
      Bf[gi].q[1] = *reinterpret_cast<const uint4*>(p + 16);
    }
#pragma unroll
    for (int ms = 0; ms < 2; ++ms)
#pragma unroll
      for (int gi = 0; gi < 4; ++gi)
        acc[ms][gi] = __builtin_amdgcn_wmma_f32_16x16x32_bf16(
            false, A[ms].v, false, Bf[gi].v, (short)0, acc[ms][gi],
            false, false);
  }

  // C/D layout: VGPR e, lanes 0-15 -> m = e ; lanes 16-31 -> m = e+8 ; n = lane%16
  const int n  = n0 + r;
  const int mh = (lane >> 4) * 8;
#pragma unroll
  for (int ms = 0; ms < 2; ++ms) {
#pragma unroll
    for (int e = 0; e < 8; ++e) {
      const int mrow = m0 + ms * 16 + mh + e;
      const int sv = sidx[mrow];
      const float* q = qtab + (size_t)sv * FH_SZ + n;
      float gi_ = acc[ms][0][e] + q[0];
      float gf_ = acc[ms][1][e] + q[1 * H_SZ];
      float gg_ = acc[ms][2][e] + q[2 * H_SZ];
      float go_ = acc[ms][3][e] + q[3 * H_SZ];
      const size_t off = (size_t)mrow * H_SZ + n;
      float cp = c_state[off];
      float c2 = sigmoidf_(gf_) * cp + sigmoidf_(gi_) * tanhf(gg_);
      float h2 = sigmoidf_(go_) * tanhf(c2);
      c_state[off]  = c2;
      h_out_f32[off] = h2;
      h_out_bf[off]  = f2bf(h2);
    }
  }
}

// ------------------------- gumbel-softmax sampling --------------------------
// One wave (32 lanes) per batch row: 3 dot products over H, softmax, argmax.
__global__ __launch_bounds__(256) void sample_kernel(
    const float* __restrict__ h, const float* __restrict__ W,
    const float* __restrict__ bias, const float* __restrict__ gu,
    float* __restrict__ vals, float* __restrict__ wbs,
    float* __restrict__ probs, int* __restrict__ sidx,
    float v0, float v1, float v2) {
  const int lane = threadIdx.x & 31;
  const int row  = blockIdx.x * 8 + (threadIdx.x >> 5);
  const float* hr = h + (size_t)row * H_SZ;
  float p0 = 0.f, p1 = 0.f, p2 = 0.f;
  for (int i = lane; i < H_SZ; i += 32) {
    float hv = hr[i];
    p0 += hv * W[i];
    p1 += hv * W[H_SZ + i];
    p2 += hv * W[2 * H_SZ + i];
  }
#pragma unroll
  for (int off = 16; off; off >>= 1) {
    p0 += __shfl_xor(p0, off, 32);
    p1 += __shfl_xor(p1, off, 32);
    p2 += __shfl_xor(p2, off, 32);
  }
  if (lane == 0) {
    float z[3];
    z[0] = p0 + bias[0];
    z[1] = p1 + bias[1];
    z[2] = p2 + bias[2];
#pragma unroll
    for (int k = 0; k < 3; ++k) {
      float u = gu[(size_t)row * 3 + k];
      z[k] += -logf(-logf(u));                 // TAU == 1.0
    }
    float m = fmaxf(z[0], fmaxf(z[1], z[2]));
    float e0 = __expf(z[0] - m), e1 = __expf(z[1] - m), e2 = __expf(z[2] - m);
    float inv = 1.0f / (e0 + e1 + e2);
    float y0 = e0 * inv, y1 = e1 * inv, y2 = e2 * inv;
    int a = 0; float best = y0;
    if (y1 > best) { best = y1; a = 1; }
    if (y2 > best) { a = 2; }
    size_t o3 = (size_t)row * 3;
    probs[o3 + 0] = y0; probs[o3 + 1] = y1; probs[o3 + 2] = y2;
    wbs[o3 + 0] = (a == 0) ? 1.f : 0.f;        // y_hard == onehot numerically
    wbs[o3 + 1] = (a == 1) ? 1.f : 0.f;
    wbs[o3 + 2] = (a == 2) ? 1.f : 0.f;
    vals[row] = (a == 0) ? v0 : ((a == 1) ? v1 : v2);
    sidx[row] = a;
  }
}

// ------------------------------- host side ----------------------------------

extern "C" void kernel_launch(void* const* d_in, const int* in_sizes, int n_in,
                              void* d_out, int out_size, void* d_ws,
                              size_t ws_size, hipStream_t stream) {
  const float* constraints = (const float*)d_in[0];
  const int*   scid        = (const int*)d_in[1];
  const float* gumbel      = (const float*)d_in[2];
  const float* cond_emb    = (const float*)d_in[3];
  const float* sc_emb      = (const float*)d_in[4];
  const float* depth_emb   = (const float*)d_in[5];
  const float* ratio_emb   = (const float*)d_in[6];
  const float* ks_emb      = (const float*)d_in[7];
  const float* W_ih        = (const float*)d_in[8];
  const float* b_ih        = (const float*)d_in[9];
  const float* W_hh        = (const float*)d_in[10];
  const float* b_hh        = (const float*)d_in[11];
  const float* depth_W     = (const float*)d_in[12];
  const float* depth_b     = (const float*)d_in[13];
  const float* width_W     = (const float*)d_in[14];
  const float* width_b     = (const float*)d_in[15];
  const float* ks_W        = (const float*)d_in[16];
  const float* ks_b        = (const float*)d_in[17];

  char* ws = (char*)d_ws;
  unsigned short* whh_bf = (unsigned short*)ws;                     // 8 MB
  unsigned short* hbfA   = (unsigned short*)(ws + ((size_t)8 << 20));  // 8 MB
  unsigned short* hbfB   = (unsigned short*)(ws + ((size_t)16 << 20)); // 8 MB
  float* h_f32 = (float*)(ws + ((size_t)24 << 20));                 // 16 MB
  float* c_f32 = (float*)(ws + ((size_t)40 << 20));                 // 16 MB
  float* Q     = (float*)(ws + ((size_t)56 << 20));                 // 9*4096
  float* C5    = Q  + 9 * FH_SZ;                                    // 5*4096
  float* Sv    = C5 + 5 * FH_SZ;                                    // 4096
  int*   sidx  = (int*)(Sv + FH_SZ);                                // 4096

  // ---- one-time (per launch) precompute ----
  convert_bf16_kernel<<<dim3((B_SZ * H_SZ) / 256), dim3(256), 0, stream>>>(
      W_hh, whh_bf);
  precompute_q_kernel<<<dim3((9 * FH_SZ) / 256), dim3(256), 0, stream>>>(
      depth_emb, ratio_emb, ks_emb, W_ih, b_ih, b_hh, Q);
  precompute_c5_kernel<<<dim3((5 * FH_SZ) / 256), dim3(256), 0, stream>>>(
      cond_emb, W_ih, C5);
  precompute_s_kernel<<<dim3(FH_SZ / 256), dim3(256), 0, stream>>>(
      sc_emb, scid, W_ih, b_ih, b_hh, Sv);

  // ---- step 0 (pure elementwise) ----
  lstm_step0_kernel<<<dim3((B_SZ * H_SZ) / 256), dim3(256), 0, stream>>>(
      constraints, C5, Sv, c_f32, h_f32, hbfA);

  // ---- output layout (floats, concatenated in return order) ----
  float* out      = (float*)d_out;
  float* depths_o = out;
  float* ratios_o = depths_o + (size_t)5 * B_SZ;
  float* kss_o    = ratios_o + (size_t)20 * B_SZ;
  float* dwbs     = kss_o + (size_t)20 * B_SZ;
  float* dprb     = dwbs + (size_t)5 * B_SZ * 3;
  float* rwbs     = dprb + (size_t)5 * B_SZ * 3;
  float* rprb     = rwbs + (size_t)20 * B_SZ * 3;
  float* kwbs     = rprb + (size_t)20 * B_SZ * 3;
  float* kprb     = kwbs + (size_t)20 * B_SZ * 3;

  // ---- step schedule: per unit [depth, (width, ks) x 4] ----
  int types[45], tix[45];
  {
    int t = 0;
    for (int u = 0; u < 5; ++u) {
      types[t] = 0; tix[t] = u; ++t;
      for (int blk = 0; blk < 4; ++blk) {
        types[t] = 1; tix[t] = u * 4 + blk; ++t;
        types[t] = 2; tix[t] = u * 4 + blk; ++t;
      }
    }
  }

  unsigned short* hin  = hbfA;
  unsigned short* hout = hbfB;
  for (int t = 0; t < 45; ++t) {
    const int ty = types[t], ix = tix[t];
    const float *W, *bb;
    float *vals, *wb, *pb;
    float v0, v1, v2;
    if (ty == 0) {
      W = depth_W; bb = depth_b;
      vals = depths_o + (size_t)ix * B_SZ;
      wb = dwbs + (size_t)ix * B_SZ * 3;
      pb = dprb + (size_t)ix * B_SZ * 3;
      v0 = 2.f; v1 = 3.f; v2 = 4.f;
    } else if (ty == 1) {
      W = width_W; bb = width_b;
      vals = ratios_o + (size_t)ix * B_SZ;
      wb = rwbs + (size_t)ix * B_SZ * 3;
      pb = rprb + (size_t)ix * B_SZ * 3;
      v0 = 3.f; v1 = 4.f; v2 = 6.f;
    } else {
      W = ks_W; bb = ks_b;
      vals = kss_o + (size_t)ix * B_SZ;
      wb = kwbs + (size_t)ix * B_SZ * 3;
      pb = kprb + (size_t)ix * B_SZ * 3;
      v0 = 3.f; v1 = 5.f; v2 = 7.f;
    }
    sample_kernel<<<dim3(B_SZ / 8), dim3(256), 0, stream>>>(
        h_f32, W, bb, gumbel + (size_t)t * B_SZ * 3, vals, wb, pb, sidx,
        v0, v1, v2);
    if (t < 44) {
      const float* qtab = Q + (size_t)ty * 3 * FH_SZ;  // embed table of step t
      lstm_step_kernel<<<dim3(1024), dim3(256), 0, stream>>>(
          hin, whh_bf, qtab, sidx, c_f32, h_f32, hout);
      unsigned short* tmp = hout; hout = hin; hin = tmp;  // ping-pong h (bf16)
    }
  }
  (void)in_sizes; (void)n_in; (void)out_size; (void)ws_size;
}
